// AdditiveAttention_43319040147615
// MI455X (gfx1250) — compile-verified, matrix operand order checked by reading, not run on hardware
//
#include <hip/hip_runtime.h>
#include <hip/hip_bf16.h>
#include <math.h>

#define NN 50000
#define EE 800000
#define DD 128
#define HH 128
#define LN_EPS 1e-3f

typedef __bf16 bf16_t;
typedef bf16_t v16bf __attribute__((ext_vector_type(16)));
typedef bf16_t v8bf  __attribute__((ext_vector_type(8)));
typedef float  v8f   __attribute__((ext_vector_type(8)));

static __device__ __forceinline__ v8f wmma_bf16(v16bf a, v16bf b, v8f c) {
  return __builtin_amdgcn_wmma_f32_16x16x32_bf16(false, a, false, b, (short)0, c,
                                                 false, false);
}

static __device__ __forceinline__ v16bf ld_frag(const bf16_t* p0, const bf16_t* p1) {
  v8bf lo = *(const v8bf*)p0;
  v8bf hi = *(const v8bf*)p1;
  return __builtin_shufflevector(lo, hi, 0, 1, 2, 3, 4, 5, 6, 7,
                                 8, 9, 10, 11, 12, 13, 14, 15);
}

static __device__ __forceinline__ void wait_async0() {
#if __has_builtin(__builtin_amdgcn_s_wait_asynccnt)
  __builtin_amdgcn_s_wait_asynccnt(0);
#else
  asm volatile("s_wait_asynccnt 0" ::: "memory");
#endif
}

// Async-copy a [128 rows][Kc cols] bf16 tile (cols starting at k0, global row
// stride Ks elems) into LDS at byte offset ldsBase, row stride Kc elems.
// Uses GLOBAL_LOAD_ASYNC_TO_LDS_B128 (GVS mode: saddr base + 32-bit voffset),
// tracked by ASYNCcnt -- data never touches VGPRs.
static __device__ __forceinline__ void stage_async(const bf16_t* __restrict__ src,
                                                   unsigned ldsBase, int Ks, int k0,
                                                   int Kc, int tid, int nthr) {
  const int cpr = Kc >> 3;  // 16B chunks per row
  const int total = 128 * cpr;
  for (int i = tid; i < total; i += nthr) {
    int nrow = i / cpr, c = i - nrow * cpr;
    unsigned lds = ldsBase + (unsigned)((nrow * Kc + c * 8) * 2);
    unsigned voff = (unsigned)(((long)nrow * Ks + k0 + c * 8) * 2);
    asm volatile("global_load_async_to_lds_b128 %0, %1, %2"
                 :: "v"(lds), "v"(voff), "s"((const void*)src)
                 : "memory");
  }
  wait_async0();
}

// Dense layer for a 16-row tile: Hout(16x128 f32) = act(A @ W^T + bias).
// A: bf16 [16][astride] LDS; Wt: bf16 [128 n][128 k] LDS.
static __device__ __forceinline__ void dense16(const bf16_t* __restrict__ A, int astride,
                                               const bf16_t* __restrict__ Wt,
                                               const float* __restrict__ bias,
                                               float* __restrict__ Hout,
                                               int l15, int klocal, int moff, bool relu) {
  for (int nt = 0; nt < 8; ++nt) {
    int n = nt * 16 + l15;
    float bv = bias[n];
    v8f acc = {bv, bv, bv, bv, bv, bv, bv, bv};
#pragma unroll
    for (int ks = 0; ks < 4; ++ks) {
      int kb = ks * 32 + klocal;
      v16bf af = ld_frag(A + l15 * astride + kb, A + l15 * astride + kb + 16);
      v16bf bf = ld_frag(Wt + n * 128 + kb, Wt + n * 128 + kb + 16);
      acc = wmma_bf16(af, bf, acc);
    }
#pragma unroll
    for (int v = 0; v < 8; ++v) {
      float z = acc[v];
      if (relu) z = fmaxf(z, 0.f);
      Hout[(moff + v) * 128 + n] = z;
    }
  }
}

// Row-wise LayerNorm of a 16x128 f32 tile (one wave, 2 lanes/row, wave32 xor-16).
static __device__ __forceinline__ void layernorm16(float* __restrict__ H,
                                                   bf16_t* __restrict__ outb, int ostride,
                                                   const float* __restrict__ g,
                                                   const float* __restrict__ be, int lane) {
  int r = lane & 15;
  int c0 = (lane >> 4) * 64;
  float s = 0.f, s2 = 0.f;
  for (int j = 0; j < 64; ++j) {
    float x = H[r * 128 + c0 + j];
    s += x;
    s2 += x * x;
  }
  s += __shfl_xor(s, 16, 32);
  s2 += __shfl_xor(s2, 16, 32);
  float mu = s * (1.f / 128.f);
  float var = s2 * (1.f / 128.f) - mu * mu;
  float rs = rsqrtf(var + LN_EPS);
  for (int j = 0; j < 64; ++j) {
    int c = c0 + j;
    float x = (H[r * 128 + c] - mu) * rs * g[c] + be[c];
    if (outb)
      outb[r * ostride + c] = (bf16_t)x;
    else
      H[r * 128 + c] = x;
  }
}

// One-time: W [K][N=128] f32 -> Wt [n][k] bf16 (n-major, stride K).
__global__ __launch_bounds__(256) void transpose_bf16_kernel(const float* __restrict__ W,
                                                             bf16_t* __restrict__ Wt,
                                                             int K) {
  int i = blockIdx.x * 256 + threadIdx.x;  // coalesced over W
  if (i < K * 128) {
    int k = i >> 7, n = i & 127;
    Wt[n * K + k] = (bf16_t)W[i];
  }
}

// ---------------------------------------------------------------------------
// memory MLP: mem_head = LN(relu(LN(relu(M@W1+b1))@W2+b2))@W3+b3   [N,128]
// 128 threads = 4 waves, 64 rows/block. Dynamic LDS: W(32K)+A(16K)+H(32K).
// ---------------------------------------------------------------------------
__global__ __launch_bounds__(128) void memory_mlp_kernel(
    const float* __restrict__ mem, const bf16_t* __restrict__ W1t,
    const bf16_t* __restrict__ W2t, const bf16_t* __restrict__ W3t,
    const float* __restrict__ b1, const float* __restrict__ b2,
    const float* __restrict__ b3, const float* __restrict__ g1,
    const float* __restrict__ be1, const float* __restrict__ g2,
    const float* __restrict__ be2, float* __restrict__ mem_head) {
  extern __shared__ __align__(16) char smem_raw[];
  bf16_t* smW = (bf16_t*)smem_raw;                 // [128][128] bf16
  bf16_t* smA = (bf16_t*)(smem_raw + 32768);       // 4 x [16][128] bf16
  float* smH = (float*)(smem_raw + 49152);         // 4 x [16][128] f32
  const unsigned smW_lds = (unsigned)(uintptr_t)smW;  // LDS aperture: addr[31:0]

  const int tid = threadIdx.x;
  const int w = tid >> 5, lane = tid & 31;
  const int l15 = lane & 15, hi = lane >> 4;
  const int klocal = hi << 3, moff = hi << 3;
  const int row0 = blockIdx.x * 64 + w * 16;
  bf16_t* myA = smA + w * 16 * 128;
  float* myH = smH + w * 16 * 128;

  // gather 16 input rows -> bf16 (float4 vectorized, 2 lanes per row)
  {
    int rg = row0 + l15;
    if (rg >= NN) rg = NN - 1;
    const float4* src = (const float4*)(mem + (long)rg * DD + hi * 64);
    bf16_t* dst = myA + l15 * 128 + hi * 64;
#pragma unroll 4
    for (int j = 0; j < 16; ++j) {
      float4 x = src[j];
      dst[j * 4 + 0] = (bf16_t)x.x;
      dst[j * 4 + 1] = (bf16_t)x.y;
      dst[j * 4 + 2] = (bf16_t)x.z;
      dst[j * 4 + 3] = (bf16_t)x.w;
    }
  }
  stage_async(W1t, smW_lds, 128, 0, 128, tid, 128);
  __syncthreads();
  dense16(myA, 128, smW, b1, myH, l15, klocal, moff, true);
  __syncthreads();
  layernorm16(myH, myA, 128, g1, be1, lane);
  stage_async(W2t, smW_lds, 128, 0, 128, tid, 128);
  __syncthreads();
  dense16(myA, 128, smW, b2, myH, l15, klocal, moff, true);
  __syncthreads();
  layernorm16(myH, myA, 128, g2, be2, lane);
  stage_async(W3t, smW_lds, 128, 0, 128, tid, 128);
  __syncthreads();
  // final layer straight to global
  for (int nt = 0; nt < 8; ++nt) {
    int n = nt * 16 + l15;
    float bv = b3[n];
    v8f acc = {bv, bv, bv, bv, bv, bv, bv, bv};
#pragma unroll
    for (int ks = 0; ks < 4; ++ks) {
      int kb = ks * 32 + klocal;
      v16bf af = ld_frag(myA + l15 * 128 + kb, myA + l15 * 128 + kb + 16);
      v16bf bf = ld_frag(smW + n * 128 + kb, smW + n * 128 + kb + 16);
      acc = wmma_bf16(af, bf, acc);
    }
#pragma unroll
    for (int v = 0; v < 8; ++v) {
      int rr = row0 + moff + v;
      if (rr < NN) mem_head[(long)rr * HH + n] = acc[v];
    }
  }
}

// ---------------------------------------------------------------------------
// edge MLP: h_e = tanh(mlp3([q[row_e] | m[col_e]]))   [E]
// 128 threads = 4 waves, 64 edges/block. LDS: W(32K)+A(32K)+H(32K).
// ---------------------------------------------------------------------------
__global__ __launch_bounds__(128) void edge_mlp_kernel(
    const float* __restrict__ query, const float* __restrict__ memv,
    const int* __restrict__ row, const int* __restrict__ col,
    const bf16_t* __restrict__ W1t, const bf16_t* __restrict__ W2t,
    const float* __restrict__ b1, const float* __restrict__ b2,
    const float* __restrict__ W3, const float* __restrict__ b3,
    const float* __restrict__ g1, const float* __restrict__ be1,
    const float* __restrict__ g2, const float* __restrict__ be2,
    float* __restrict__ h_out) {
  extern __shared__ __align__(16) char smem_raw[];
  bf16_t* smW = (bf16_t*)smem_raw;                 // [128][128] bf16
  bf16_t* smA = (bf16_t*)(smem_raw + 32768);       // 4 x [16][256] bf16
  float* smH = (float*)(smem_raw + 65536);         // 4 x [16][128] f32
  const unsigned smW_lds = (unsigned)(uintptr_t)smW;

  const int tid = threadIdx.x;
  const int w = tid >> 5, lane = tid & 31;
  const int l15 = lane & 15, hi = lane >> 4;
  const int klocal = hi << 3, moff = hi << 3;
  const int e0 = blockIdx.x * 64 + w * 16;
  bf16_t* myA = smA + w * 16 * 256;
  float* myH = smH + w * 16 * 128;

  // gather: cols 0..127 = query[row[e]], 128..255 = memory[col[e]]
  {
    int e = e0 + l15;
    if (e >= EE) e = EE - 1;
    const float4* src =
        (const float4*)(hi ? (memv + (long)col[e] * DD) : (query + (long)row[e] * DD));
    bf16_t* dst = myA + l15 * 256 + hi * 128;
#pragma unroll 4
    for (int j = 0; j < 32; ++j) {
      float4 x = src[j];
      dst[j * 4 + 0] = (bf16_t)x.x;
      dst[j * 4 + 1] = (bf16_t)x.y;
      dst[j * 4 + 2] = (bf16_t)x.z;
      dst[j * 4 + 3] = (bf16_t)x.w;
    }
  }

  // layer 1: K=256 in two staged 128-K halves, accumulators live across both
  v8f acc[8];
#pragma unroll
  for (int nt = 0; nt < 8; ++nt) {
    float bv = b1[nt * 16 + l15];
    acc[nt] = (v8f){bv, bv, bv, bv, bv, bv, bv, bv};
  }
  for (int kh = 0; kh < 2; ++kh) {
    stage_async(W1t, smW_lds, 256, kh * 128, 128, tid, 128);
    __syncthreads();
    for (int nt = 0; nt < 8; ++nt) {
      int n = nt * 16 + l15;
#pragma unroll
      for (int ks = 0; ks < 4; ++ks) {
        int ka = kh * 128 + ks * 32 + klocal;
        int kb = ks * 32 + klocal;
        v16bf af = ld_frag(myA + l15 * 256 + ka, myA + l15 * 256 + ka + 16);
        v16bf bf = ld_frag(smW + n * 128 + kb, smW + n * 128 + kb + 16);
        acc[nt] = wmma_bf16(af, bf, acc[nt]);
      }
    }
    __syncthreads();
  }
#pragma unroll
  for (int nt = 0; nt < 8; ++nt)
#pragma unroll
    for (int v = 0; v < 8; ++v)
      myH[(moff + v) * 128 + nt * 16 + l15] = fmaxf(acc[nt][v], 0.f);
  __syncthreads();
  layernorm16(myH, myA, 256, g1, be1, lane);
  stage_async(W2t, smW_lds, 128, 0, 128, tid, 128);
  __syncthreads();
  dense16(myA, 256, smW, b2, myH, l15, klocal, moff, true);
  __syncthreads();
  layernorm16(myH, nullptr, 0, g2, be2, lane);  // keep f32 for final dot
  __syncthreads();

  // layer 3: H -> 1 dot, tanh
  {
    float part = 0.f;
    int c0 = hi * 64;
    for (int j = 0; j < 64; ++j) part += myH[l15 * 128 + c0 + j] * W3[c0 + j];
    part += __shfl_xor(part, 16, 32);
    if (hi == 0) {
      int e = e0 + l15;
      if (e < EE) h_out[e] = tanhf(part + b3[0]);
    }
  }
}

// ---------------------------------------------------------------------------
// scatter: out[row_e] += h_e * mem_head[col_e]; row sorted -> run-length
// accumulate in registers, atomic flush on row change.
// ---------------------------------------------------------------------------
__global__ __launch_bounds__(128) void scatter_kernel(const int* __restrict__ row,
                                                      const int* __restrict__ col,
                                                      const float* __restrict__ h,
                                                      const float* __restrict__ mem_head,
                                                      float* __restrict__ out) {
  const int f = threadIdx.x;
  int e0 = blockIdx.x * 64;
  int e1 = e0 + 64;
  if (e1 > EE) e1 = EE;
  if (e0 >= e1) return;
  int cur = row[e0];
  float acc = 0.f;
  for (int e = e0; e < e1; ++e) {
    int r = row[e];
    if (r != cur) {
      atomicAdd(&out[(long)cur * HH + f], acc);
      acc = 0.f;
      cur = r;
    }
    acc = fmaf(h[e], mem_head[(long)col[e] * HH + f], acc);
  }
  atomicAdd(&out[(long)cur * HH + f], acc);
}

__global__ void zero_kernel(float* __restrict__ p, long n) {
  long i = (long)blockIdx.x * blockDim.x + threadIdx.x;
  if (i < n) p[i] = 0.f;
}

extern "C" void kernel_launch(void* const* d_in, const int* in_sizes, int n_in,
                              void* d_out, int out_size, void* d_ws, size_t ws_size,
                              hipStream_t stream) {
  const float* query = (const float*)d_in[0];
  const float* memv = (const float*)d_in[1];
  const int* row = (const int*)d_in[2];
  const int* col = (const int*)d_in[3];
  const float* pW1 = (const float*)d_in[4];
  const float* pb1 = (const float*)d_in[5];
  const float* pW2 = (const float*)d_in[6];
  const float* pb2 = (const float*)d_in[7];
  const float* pW3 = (const float*)d_in[8];
  const float* pb3 = (const float*)d_in[9];
  const float* pg1 = (const float*)d_in[10];
  const float* pbe1 = (const float*)d_in[11];
  const float* pg2 = (const float*)d_in[12];
  const float* pbe2 = (const float*)d_in[13];
  const float* mW1 = (const float*)d_in[14];
  const float* mb1 = (const float*)d_in[15];
  const float* mW2 = (const float*)d_in[16];
  const float* mb2 = (const float*)d_in[17];
  const float* mW3 = (const float*)d_in[18];
  const float* mb3 = (const float*)d_in[19];
  const float* mg1 = (const float*)d_in[20];
  const float* mbe1 = (const float*)d_in[21];
  const float* mg2 = (const float*)d_in[22];
  const float* mbe2 = (const float*)d_in[23];

  float* mem_head = (float*)d_ws;            // [N,H] f32
  float* h_edge = mem_head + (long)NN * HH;  // [E]   f32
  bf16_t* wbuf = (bf16_t*)(h_edge + EE);     // pre-converted transposed weights
  bf16_t* pW1t = wbuf;                       // [128][256]
  bf16_t* pW2t = wbuf + 32768;               // [128][128]
  bf16_t* mW1t = wbuf + 49152;
  bf16_t* mW2t = wbuf + 65536;
  bf16_t* mW3t = wbuf + 81920;
  float* out = (float*)d_out;

  zero_kernel<<<((long)NN * HH + 255) / 256, 256, 0, stream>>>(out, (long)NN * HH);

  transpose_bf16_kernel<<<(256 * 128 + 255) / 256, 256, 0, stream>>>(pW1, pW1t, 256);
  transpose_bf16_kernel<<<(128 * 128 + 255) / 256, 256, 0, stream>>>(pW2, pW2t, 128);
  transpose_bf16_kernel<<<(128 * 128 + 255) / 256, 256, 0, stream>>>(mW1, mW1t, 128);
  transpose_bf16_kernel<<<(128 * 128 + 255) / 256, 256, 0, stream>>>(mW2, mW2t, 128);
  transpose_bf16_kernel<<<(128 * 128 + 255) / 256, 256, 0, stream>>>(mW3, mW3t, 128);

  memory_mlp_kernel<<<(NN + 63) / 64, 128, 81920, stream>>>(
      memv, mW1t, mW2t, mW3t, mb1, mb2, mb3, mg1, mbe1, mg2, mbe2, mem_head);

  edge_mlp_kernel<<<(EE + 63) / 64, 128, 98304, stream>>>(
      query, memv, row, col, pW1t, pW2t, pb1, pb2, pW3, pb3, pg1, pbe1, pg2, pbe2,
      h_edge);

  scatter_kernel<<<(EE + 63) / 64, 128, 0, stream>>>(row, col, h_edge, mem_head, out);
}
